// MyClassificationGCN_25013889532261
// MI455X (gfx1250) — compile-verified, matrix operand order checked by reading, not run on hardware
//
#include <hip/hip_runtime.h>
#include <hip/hip_bf16.h>

// ---------------- types ----------------
typedef __attribute__((ext_vector_type(16))) _Float16 v16h;
typedef __attribute__((ext_vector_type(8)))  float    v8f;

// ---------------- problem constants (match reference) ----------------
#define NNODE   50000
#define NEDGE   800000
#define F_IN    1433
#define HID     256
#define NCLASS  7

#define MPAD    50048          // 3128 * 16
#define MTILES  3128
#define K0PAD   1440           // 45 * 32
#define NKB0    45
#define NKB1    8              // 256 / 32

static __device__ __forceinline__ unsigned short f2h(float x) {
    _Float16 h = (_Float16)x;
    return __builtin_bit_cast(unsigned short, h);
}

// ---------- fp32 -> f16 padded row-major activation convert ----------
__global__ void convert_a_f16(const float* __restrict__ src, unsigned short* __restrict__ dst,
                              int Mreal, int Kreal, int ldSrc, int Kp, long long total) {
    long long idx = (long long)blockIdx.x * blockDim.x + threadIdx.x;
    if (idx >= total) return;
    int k   = (int)(idx % Kp);
    int row = (int)(idx / Kp);
    unsigned short v = 0;
    if (row < Mreal && k < Kreal) v = f2h(src[(long long)row * ldSrc + k]);
    dst[idx] = v;
}

// ---------- weight pack into WMMA B-fragment order ----------
// For k-block kb (32 K), n-tile nt (16 N): 512 f16 chunk, lane-major.
// lane<16: col=lane, K = kb+{0..7,16..23}; lane>=16: col=lane-16, K = kb+{8..15,24..31}.
__global__ void pack_w_f16(const float* __restrict__ W, unsigned short* __restrict__ dst,
                           int Kreal, int Nreal, int nKB, int NT, int total) {
    int idx = blockIdx.x * blockDim.x + threadIdx.x;
    if (idx >= total) return;
    int j     = idx & 15;
    int lane  = (idx >> 4) & 31;
    int blk   = idx >> 9;            // kb * NT + nt
    int nt    = blk % NT;
    int kb    = blk / NT;
    int khalf = lane >> 4;
    int k_off = (j < 8) ? (8 * khalf + j) : (16 + 8 * khalf + (j - 8));
    int n = nt * 16 + (lane & 15);
    int k = kb * 32 + k_off;
    unsigned short v = 0;
    if (k < Kreal && n < Nreal) v = f2h(W[(long long)k * Nreal + n]);
    dst[idx] = v;
}

// ---------- WMMA GEMM: C[MPAD x Nld] = A(f16) @ Bpack(f16), f32 accum ----------
// 1 wave -> 16 rows x (TILES*16) cols. Branchless, double-buffered K loop.
template <int TILES>
__global__ void gemm_wmma_f16(const unsigned short* __restrict__ A,
                              const unsigned short* __restrict__ Bpack,
                              float* __restrict__ C,
                              int nKB, int Kp, int NT, int nGroups, int Nld) {
    int wid  = (int)((blockIdx.x * blockDim.x + threadIdx.x) >> 5);
    int lane = threadIdx.x & 31;
    int totalW = MTILES * nGroups;
    if (wid >= totalW) return;        // grids are exact; never partial-EXEC
    int ntg = wid % nGroups;
    int mt  = wid / nGroups;

    int khalf = lane >> 4;
    int rloc  = lane & 15;
    const unsigned short* arow  = A + (long long)(mt * 16 + rloc) * Kp + 8 * khalf;
    const unsigned short* bbase = Bpack + (long long)(ntg * TILES) * 512 + lane * 16;
    const long long bKstride = (long long)NT * 512;

    v8f z = {0.f, 0.f, 0.f, 0.f, 0.f, 0.f, 0.f, 0.f};
    v8f acc[TILES];
#pragma unroll
    for (int t = 0; t < TILES; ++t) acc[t] = z;

    // ---- prologue: load kb = 0 fragments ----
    v16h a_cur, b_cur[TILES];
    *((uint4*)&a_cur)     = *(const uint4*)(arow);
    *((uint4*)&a_cur + 1) = *(const uint4*)(arow + 16);
#pragma unroll
    for (int t = 0; t < TILES; ++t)
        b_cur[t] = *(const v16h*)(bbase + (long long)t * 512);

    for (int kb = 0; kb < nKB - 1; ++kb) {
        // load next iteration's fragments (overlaps with WMMA issue below)
        v16h a_nxt, b_nxt[TILES];
        *((uint4*)&a_nxt)     = *(const uint4*)(arow + (kb + 1) * 32);
        *((uint4*)&a_nxt + 1) = *(const uint4*)(arow + (kb + 1) * 32 + 16);
#pragma unroll
        for (int t = 0; t < TILES; ++t)
            b_nxt[t] = *(const v16h*)(bbase + (long long)(kb + 1) * bKstride + (long long)t * 512);
        // speculative prefetch 2 K-blocks ahead (guard-free: address stays in
        // the padded workspace, and CDNA5 prefetch drops failed translations)
        __builtin_prefetch(arow + (kb + 2) * 32, 0, 3);

#pragma unroll
        for (int t = 0; t < TILES; ++t)
            acc[t] = __builtin_amdgcn_wmma_f32_16x16x32_f16(
                false, a_cur, false, b_cur[t], (short)0, acc[t], false, false);

        a_cur = a_nxt;
#pragma unroll
        for (int t = 0; t < TILES; ++t) b_cur[t] = b_nxt[t];
    }
    // ---- epilogue: last K block ----
#pragma unroll
    for (int t = 0; t < TILES; ++t)
        acc[t] = __builtin_amdgcn_wmma_f32_16x16x32_f16(
            false, a_cur, false, b_cur[t], (short)0, acc[t], false, false);

    // store: VGPR r -> row mt*16 + r + 8*khalf, col (ntg*TILES+t)*16 + (lane&15)
#pragma unroll
    for (int t = 0; t < TILES; ++t) {
        float* cb = C + (long long)(mt * 16 + 8 * khalf) * Nld + (ntg * TILES + t) * 16 + rloc;
#pragma unroll
        for (int r = 0; r < 8; ++r) cb[(long long)r * Nld] = acc[t][r];
    }
}

// ---------- fill out with bias (padded cols -> 0) ----------
__global__ void fill_bias(float* __restrict__ out, const float* __restrict__ bias,
                          int C, int ld, long long total) {
    long long idx = (long long)blockIdx.x * blockDim.x + threadIdx.x;
    if (idx >= total) return;
    int c = (int)(idx % ld);
    out[idx] = (c < C) ? bias[c] : 0.f;
}

// ---------- edge-parallel SpMM scatter-add (wave per edge) ----------
__global__ void spmm_atomic(const float* __restrict__ sup, const int* __restrict__ rows,
                            const int* __restrict__ cols, const float* __restrict__ vals,
                            float* __restrict__ out, int E, int C, int ld) {
    int wid  = (int)((blockIdx.x * blockDim.x + threadIdx.x) >> 5);
    int lane = threadIdx.x & 31;
    int nW   = (int)((gridDim.x * blockDim.x) >> 5);
    for (int e = wid; e < E; e += nW) {
        int dr = rows[e];
        int sr = cols[e];
        float v = vals[e];
        const float* s = sup + (long long)sr * ld;
        float* o = out + (long long)dr * ld;
        for (int c = lane; c < C; c += 32)
            atomicAdd(&o[c], v * s[c]);
    }
}

__global__ void zerov(float* p, int n) {
    int i = blockIdx.x * blockDim.x + threadIdx.x;
    if (i < n) p[i] = 0.f;
}

// ---------- column sums (for PairNorm mean), optional ReLU ----------
__global__ void colsum_k(const float* __restrict__ h, float* __restrict__ sums,
                         int Nrows, int C, int ld, int rowsPerBlock, int doRelu) {
    int c = threadIdx.x;
    if (c >= C) return;
    int r0 = blockIdx.x * rowsPerBlock;
    int r1 = r0 + rowsPerBlock; if (r1 > Nrows) r1 = Nrows;
    float s = 0.f;
    for (int r = r0; r < r1; ++r) {
        float x = h[(long long)r * ld + c];
        if (doRelu) x = fmaxf(x, 0.f);
        s += x;
    }
    atomicAdd(&sums[c], s);
}

// ---------- fused (ReLU) -> center -> row normalize -> (sigmoid) ----------
__global__ void rownorm_k(const float* __restrict__ h, const float* __restrict__ sums,
                          float* __restrict__ dst, int Nrows, int C, int ldSrc, int ldDst,
                          float invN, int doRelu, int doSigmoid) {
    int wid  = (int)((blockIdx.x * blockDim.x + threadIdx.x) >> 5);
    int lane = threadIdx.x & 31;
    if (wid >= Nrows) return;
    float v[8];
    int   cs[8];
    int cnt = 0;
    float ss = 0.f;
    for (int c = lane; c < C; c += 32) {
        float x = h[(long long)wid * ldSrc + c];
        if (doRelu) x = fmaxf(x, 0.f);
        x -= sums[c] * invN;
        v[cnt] = x; cs[cnt] = c; ++cnt;
        ss += x * x;
    }
#pragma unroll
    for (int o = 16; o > 0; o >>= 1) ss += __shfl_xor(ss, o, 32);
    float rn = rsqrtf(1e-6f + ss);
    for (int i = 0; i < cnt; ++i) {
        float y = v[i] * rn;
        if (doSigmoid) y = 1.f / (1.f + __expf(-y));
        dst[(long long)wid * ldDst + cs[i]] = y;
    }
}

// ---------------- host orchestration ----------------
static inline size_t alignup(size_t x) { return (x + 255) & ~(size_t)255; }

extern "C" void kernel_launch(void* const* d_in, const int* in_sizes, int n_in,
                              void* d_out, int out_size, void* d_ws, size_t ws_size,
                              hipStream_t stream) {
    (void)in_sizes; (void)n_in; (void)out_size; (void)ws_size;

    const float* x   = (const float*)d_in[0];
    const int*   er  = (const int*)d_in[1];
    const int*   ec  = (const int*)d_in[2];
    const float* ev  = (const float*)d_in[3];
    const float* W0  = (const float*)d_in[4];
    const float* b0  = (const float*)d_in[5];
    const float* W1  = (const float*)d_in[6];
    const float* b1  = (const float*)d_in[7];
    const float* W2  = (const float*)d_in[8];
    const float* b2  = (const float*)d_in[9];
    float* out = (float*)d_out;

    // workspace layout
    char* ws = (char*)d_ws;
    size_t off = 0;
    unsigned short* Ah = (unsigned short*)(ws + off); off = alignup(off + (size_t)MPAD * K0PAD * 2);
    unsigned short* Bh = (unsigned short*)(ws + off); off = alignup(off + (size_t)K0PAD * HID * 2);
    float* sup = (float*)(ws + off); off = alignup(off + (size_t)MPAD * HID * 4);
    float* h   = (float*)(ws + off); off = alignup(off + (size_t)MPAD * HID * 4);
    float* csum = (float*)(ws + off); off = alignup(off + 256 * 4);

    const int TB = 256;
    const float invN = 1.0f / (float)NNODE;

    // ================= Layer 0: x[50000x1433] @ W0[1433x256] =================
    {
        long long totA = (long long)MPAD * K0PAD;
        convert_a_f16<<<(unsigned)((totA + TB - 1) / TB), TB, 0, stream>>>(
            x, Ah, NNODE, F_IN, F_IN, K0PAD, totA);
        int totW = NKB0 * (HID / 16) * 512;
        pack_w_f16<<<(totW + TB - 1) / TB, TB, 0, stream>>>(W0, Bh, F_IN, HID, NKB0, HID / 16, totW);
        int nGroups = (HID / 16) / 4;                       // 4
        int waves = MTILES * nGroups;                       // 12512 -> exact blocks
        gemm_wmma_f16<4><<<(waves * 32) / TB, TB, 0, stream>>>(
            Ah, Bh, sup, NKB0, K0PAD, HID / 16, nGroups, HID);
        long long totH = (long long)MPAD * HID;
        fill_bias<<<(unsigned)((totH + TB - 1) / TB), TB, 0, stream>>>(h, b0, HID, HID, totH);
        spmm_atomic<<<2048, TB, 0, stream>>>(sup, er, ec, ev, h, NEDGE, HID, HID);
        zerov<<<1, TB, 0, stream>>>(csum, 256);
        colsum_k<<<(NNODE + 255) / 256, TB, 0, stream>>>(h, csum, NNODE, HID, HID, 256, 1);
        rownorm_k<<<(NNODE * 32 + TB - 1) / TB, TB, 0, stream>>>(
            h, csum, sup, NNODE, HID, HID, HID, invN, 1, 0);   // sup <- pairnorm(relu(h))
    }

    // ================= Layer 1: sup[50000x256] @ W1[256x256] =================
    {
        long long totA = (long long)MPAD * HID;
        convert_a_f16<<<(unsigned)((totA + TB - 1) / TB), TB, 0, stream>>>(
            sup, Ah, NNODE, HID, HID, HID, totA);
        int totW = NKB1 * (HID / 16) * 512;
        pack_w_f16<<<(totW + TB - 1) / TB, TB, 0, stream>>>(W1, Bh, HID, HID, NKB1, HID / 16, totW);
        int nGroups = (HID / 16) / 4;
        int waves = MTILES * nGroups;
        gemm_wmma_f16<4><<<(waves * 32) / TB, TB, 0, stream>>>(
            Ah, Bh, sup, NKB1, HID, HID / 16, nGroups, HID);
        long long totH = (long long)MPAD * HID;
        fill_bias<<<(unsigned)((totH + TB - 1) / TB), TB, 0, stream>>>(h, b1, HID, HID, totH);
        spmm_atomic<<<2048, TB, 0, stream>>>(sup, er, ec, ev, h, NEDGE, HID, HID);
        zerov<<<1, TB, 0, stream>>>(csum, 256);
        colsum_k<<<(NNODE + 255) / 256, TB, 0, stream>>>(h, csum, NNODE, HID, HID, 256, 1);
        rownorm_k<<<(NNODE * 32 + TB - 1) / TB, TB, 0, stream>>>(
            h, csum, sup, NNODE, HID, HID, HID, invN, 1, 0);
    }

    // ================= Layer 2: sup[50000x256] @ W2[256x7] (N padded to 16) =================
    {
        const int NP = 16;
        long long totA = (long long)MPAD * HID;
        convert_a_f16<<<(unsigned)((totA + TB - 1) / TB), TB, 0, stream>>>(
            sup, Ah, NNODE, HID, HID, HID, totA);
        int totW = NKB1 * 1 * 512;
        pack_w_f16<<<(totW + TB - 1) / TB, TB, 0, stream>>>(W2, Bh, HID, NCLASS, NKB1, 1, totW);
        int waves = MTILES;                                 // nGroups = 1, NT = 1 -> 391 blocks exact
        gemm_wmma_f16<1><<<(waves * 32) / TB, TB, 0, stream>>>(
            Ah, Bh, h, NKB1, HID, 1, 1, NP);                // h <- support (ld 16)
        long long totS = (long long)MPAD * NP;
        fill_bias<<<(unsigned)((totS + TB - 1) / TB), TB, 0, stream>>>(sup, b2, NCLASS, NP, totS);
        spmm_atomic<<<2048, TB, 0, stream>>>(h, er, ec, ev, sup, NEDGE, NCLASS, NP);
        zerov<<<1, TB, 0, stream>>>(csum, 256);
        colsum_k<<<(NNODE + 255) / 256, TB, 0, stream>>>(sup, csum, NNODE, NCLASS, NP, 256, 0);
        rownorm_k<<<(NNODE * 32 + TB - 1) / TB, TB, 0, stream>>>(
            sup, csum, out, NNODE, NCLASS, NP, NCLASS, invN, 0, 1);  // sigmoid -> d_out
    }
}
